// MultiHeadAttention_3006477107485
// MI455X (gfx1250) — matrix operand order repaired and checked by reading, not compiled, not run on hardware
//
#include <hip/hip_runtime.h>
#include <hip/hip_bf16.h>

#define BB 4
#define TT 2048
#define DD 1024
#define HH 16
#define HD 64
#define MM (BB * TT)   // 8192 rows

typedef __attribute__((ext_vector_type(16))) __bf16          v16bf;
typedef __attribute__((ext_vector_type(8)))  float           v8f;
typedef __attribute__((ext_vector_type(8)))  unsigned short  u16x8;
typedef __attribute__((ext_vector_type(4)))  unsigned short  u16x4;

union BFrag { v16bf v; u16x8 h[2]; unsigned short u[16]; };

// float -> bf16 (round-to-nearest-even), stored as ushort
__device__ __forceinline__ unsigned short f2bfu(float f) {
  union { float f; unsigned u; } a; a.f = f;
  unsigned r = a.u + 0x7FFFu + ((a.u >> 16) & 1u);
  return (unsigned short)(r >> 16);
}

// ---------------------------------------------------------------------------
// Pre-pass A: elementwise f32 -> bf16 (x activations). float4 in, u16x4 out.
// ---------------------------------------------------------------------------
__global__ __launch_bounds__(256) void convert_bf16(
    const float* __restrict__ in, unsigned short* __restrict__ out)
{
  const size_t i = (size_t)(blockIdx.x * 256 + threadIdx.x) * 4;
  const float4 v4 = *reinterpret_cast<const float4*>(&in[i]);
  u16x4 pk;
  pk[0] = f2bfu(v4.x); pk[1] = f2bfu(v4.y);
  pk[2] = f2bfu(v4.z); pk[3] = f2bfu(v4.w);
  *reinterpret_cast<u16x4*>(&out[i]) = pk;
}

// ---------------------------------------------------------------------------
// Pre-pass B: weight transpose + convert: out[n][k] = bf16(in[k][n]).
// 32x32 tile through LDS, block = 256 (32x8).
// ---------------------------------------------------------------------------
__global__ __launch_bounds__(256) void transpose_bf16(
    const float* __restrict__ in, unsigned short* __restrict__ out)
{
  __shared__ unsigned short t[32][34];
  const int bx = blockIdx.x * 32, by = blockIdx.y * 32;
  const int tx = threadIdx.x & 31, ty = threadIdx.x >> 5;
#pragma unroll
  for (int i = 0; i < 4; ++i)
    t[ty + 8 * i][tx] = f2bfu(in[(size_t)(by + ty + 8 * i) * DD + bx + tx]);
  __syncthreads();
#pragma unroll
  for (int i = 0; i < 4; ++i)
    out[(size_t)(bx + ty + 8 * i) * DD + by + tx] = t[tx][ty + 8 * i];
}

// ---------------------------------------------------------------------------
// Kernel 1: Y = x @ W + b (bf16 inputs, W pre-transposed), optional RoPE.
//   doRope=1: write bf16 [B,H,T,HD]  (q and k)
//   doRope=0: write bf16 [B,H,HD,T]  (v, transposed for contiguous V-frags)
// No LDS: A and B fragments are contiguous global b128 loads straight from
// the L2-resident bf16 copies. Block = 128 thr (4 indep waves), block tile
// 64x64, wave tile 16x64. Inner loop: 10 b128 loads + 4 wmma.
// ---------------------------------------------------------------------------
__global__ __launch_bounds__(128) void qkv_rope_gemm(
    const unsigned short* __restrict__ xbf, const unsigned short* __restrict__ Wt,
    const float* __restrict__ bias, const float* __restrict__ cosT,
    const float* __restrict__ sinT, unsigned short* __restrict__ outws,
    int doRope)
{
  const int tid = threadIdx.x;
  const int wv = tid >> 5, ln = tid & 31;
  const int h2 = ln >> 4, nc = ln & 15;
  const int tileN = blockIdx.x * 64, tileM = blockIdx.y * 64;

  const unsigned short* arow = xbf + (size_t)(tileM + 16 * wv + nc) * DD;
  const unsigned short* brow0 = Wt + (size_t)(tileN + nc) * DD + 16 * h2;

  v8f acc[4] = {};

#pragma unroll 1
  for (int k0 = 0; k0 < DD; k0 += 32) {
    BFrag af;                                // A: rows contiguous in k
    af.h[0] = *reinterpret_cast<const u16x8*>(&arow[k0 + 16 * h2]);
    af.h[1] = *reinterpret_cast<const u16x8*>(&arow[k0 + 16 * h2 + 8]);
#pragma unroll
    for (int f = 0; f < 4; ++f) {            // B: Wt rows contiguous in k
      const unsigned short* b = brow0 + (size_t)(16 * f) * DD + k0;
      BFrag bfr;
      bfr.h[0] = *reinterpret_cast<const u16x8*>(b);
      bfr.h[1] = *reinterpret_cast<const u16x8*>(b + 8);
      acc[f] = __builtin_amdgcn_wmma_f32_16x16x32_bf16(
          false, af.v, false, bfr.v, (short)0, acc[f], false, false);
    }
  }

  // bias
#pragma unroll
  for (int f = 0; f < 4; ++f) {
    const float bv = bias[tileN + 16 * f + nc];
#pragma unroll
    for (int r = 0; r < 8; ++r) acc[f][r] += bv;
  }

  if (doRope) {
    // RoPE in registers: col d partner (d+32) is fragment f+2, same lane
#pragma unroll
    for (int f = 0; f < 2; ++f) {
      const int d0 = 16 * f + nc;   // 0..31
      const int d1 = d0 + 32;
#pragma unroll
      for (int r = 0; r < 8; ++r) {
        const int m = tileM + 16 * wv + r + 8 * h2;
        const int t = m & (TT - 1);
        const float c0 = cosT[t * HD + d0], s0 = sinT[t * HD + d0];
        const float c1 = cosT[t * HD + d1], s1 = sinT[t * HD + d1];
        const float a = acc[f][r], b = acc[f + 2][r];
        acc[f][r]     = a * c0 - b * s0;      // rot_half low  = -x[d+32]
        acc[f + 2][r] = b * c1 + a * s1;      // rot_half high = +x[d]
      }
    }
    // store bf16 [B,H,T,HD]
#pragma unroll
    for (int f = 0; f < 4; ++f) {
      const int n = tileN + 16 * f + nc;
      const int h = n >> 6, d = n & 63;
#pragma unroll
      for (int r = 0; r < 8; ++r) {
        const int m = tileM + 16 * wv + r + 8 * h2;
        const int b = m >> 11, t = m & (TT - 1);
        outws[((size_t)((b * HH + h) * TT + t)) * HD + d] = f2bfu(acc[f][r]);
      }
    }
  } else {
    // V path: store transposed bf16 [B,H,HD,T]
#pragma unroll
    for (int f = 0; f < 4; ++f) {
      const int n = tileN + 16 * f + nc;
      const int h = n >> 6, d = n & 63;
#pragma unroll
      for (int r = 0; r < 8; ++r) {
        const int m = tileM + 16 * wv + r + 8 * h2;
        const int b = m >> 11, t = m & (TT - 1);
        outws[((size_t)((b * HH + h) * HD + d)) * TT + t] = f2bfu(acc[f][r]);
      }
    }
  }
}

// ---------------------------------------------------------------------------
// Kernel 2: causal flash attention. Block = 128 thr; wave owns 16 q-rows.
// Per 32-key tile: S=QK^T (4 wmma), online softmax, O+=PV (4 wmma).
// Q/K frags: contiguous b128 loads from [B,H,T,HD]; V frags: contiguous
// b128 loads from transposed [B,H,HD,T]. P transposed C-layout -> A-layout
// via per-wave LDS bounce ordered by s_wait_dscnt 0 (waves have divergent
// causal trip counts, so no WG barrier is possible/needed).
// ---------------------------------------------------------------------------
__global__ __launch_bounds__(128) void flash_attn(
    const unsigned short* __restrict__ q, const unsigned short* __restrict__ k,
    const unsigned short* __restrict__ v, unsigned short* __restrict__ o)
{
  __shared__ __align__(16) unsigned short pLDS[4][16][40];
  const int tid = threadIdx.x;
  const int wv = tid >> 5, ln = tid & 31;
  const int h2 = ln >> 4, nc = ln & 15;
  const int bh = blockIdx.y;
  const int qBase = blockIdx.x * 64 + 16 * wv;

  const unsigned short* qb = q + (size_t)bh * TT * HD;
  const unsigned short* kb = k + (size_t)bh * TT * HD;
  const unsigned short* vb = v + (size_t)bh * HD * TT;  // transposed layout

  BFrag qf[2];                       // Q 16x64 = two 16x32 A-frags
  {
    const int tq = qBase + nc;
#pragma unroll
    for (int half = 0; half < 2; ++half) {
      qf[half].h[0] = *reinterpret_cast<const u16x8*>(
          &qb[tq * HD + 32 * half + 16 * h2]);
      qf[half].h[1] = *reinterpret_cast<const u16x8*>(
          &qb[tq * HD + 32 * half + 16 * h2 + 8]);
    }
  }

  v8f O[4] = {};
  float mrow[8], lrow[8];
#pragma unroll
  for (int r = 0; r < 8; ++r) { mrow[r] = -1e30f; lrow[r] = 0.0f; }

  const int jEnd = qBase + 16;       // causal bound for this wave's rows
#pragma unroll 1
  for (int j0 = 0; j0 < jEnd; j0 += 32) {
    v8f S[2] = {};
#pragma unroll
    for (int c = 0; c < 2; ++c) {    // two 16-key column groups
      const int j = j0 + 16 * c + nc;
#pragma unroll
      for (int half = 0; half < 2; ++half) {
        BFrag kf;                    // B frag: contiguous in K row
        kf.h[0] = *reinterpret_cast<const u16x8*>(
            &kb[j * HD + 32 * half + 16 * h2]);
        kf.h[1] = *reinterpret_cast<const u16x8*>(
            &kb[j * HD + 32 * half + 16 * h2 + 8]);
        S[c] = __builtin_amdgcn_wmma_f32_16x16x32_bf16(
            false, qf[half].v, false, kf.v, (short)0, S[c], false, false);
      }
    }

    float alpha[8];
#pragma unroll
    for (int r = 0; r < 8; ++r) {
      const int m = qBase + r + 8 * h2;
      float sa = S[0][r] * 0.125f;   // scale = HD^-0.5
      float sb = S[1][r] * 0.125f;
      if (j0 + nc > m)      sa = -1e30f;   // causal mask
      if (j0 + 16 + nc > m) sb = -1e30f;
      float mx = fmaxf(sa, sb);
#pragma unroll
      for (int off = 8; off >= 1; off >>= 1)
        mx = fmaxf(mx, __shfl_xor(mx, off, 16));
      const float mnew = fmaxf(mrow[r], mx);
      const float al = __expf(mrow[r] - mnew);
      const float pa = __expf(sa - mnew);
      const float pb = __expf(sb - mnew);
      float ps = pa + pb;
#pragma unroll
      for (int off = 8; off >= 1; off >>= 1)
        ps += __shfl_xor(ps, off, 16);
      lrow[r] = lrow[r] * al + ps;
      mrow[r] = mnew;
      alpha[r] = al;
      pLDS[wv][r + 8 * h2][nc]      = f2bfu(pa);
      pLDS[wv][r + 8 * h2][16 + nc] = f2bfu(pb);
    }

#pragma unroll
    for (int f = 0; f < 4; ++f)
#pragma unroll
      for (int r = 0; r < 8; ++r)
        O[f][r] *= alpha[r];

    asm volatile("s_wait_dscnt 0" ::: "memory");  // order LDS write->read

    BFrag pf;                        // P as A-frag: two aligned b128 loads
    pf.h[0] = *reinterpret_cast<const u16x8*>(&pLDS[wv][nc][16 * h2]);
    pf.h[1] = *reinterpret_cast<const u16x8*>(&pLDS[wv][nc][16 * h2 + 8]);

#pragma unroll
    for (int f = 0; f < 4; ++f) {    // V B-frags contiguous in [HD,T] layout
      BFrag vf;
      const unsigned short* vrow = &vb[(size_t)(16 * f + nc) * TT + j0 + 16 * h2];
      vf.h[0] = *reinterpret_cast<const u16x8*>(vrow);
      vf.h[1] = *reinterpret_cast<const u16x8*>(vrow + 8);
      O[f] = __builtin_amdgcn_wmma_f32_16x16x32_bf16(
          false, pf.v, false, vf.v, (short)0, O[f], false, false);
    }
  }

  // normalize (8 reciprocals, not 32 divides) and write bf16 [B,T,D]
  float inv[8];
#pragma unroll
  for (int r = 0; r < 8; ++r) inv[r] = 1.0f / lrow[r];
  const int b = bh >> 4, h = bh & 15;
#pragma unroll
  for (int f = 0; f < 4; ++f) {
    const int dcol = h * HD + 16 * f + nc;
#pragma unroll
    for (int r = 0; r < 8; ++r) {
      const int t = qBase + r + 8 * h2;
      o[((size_t)(b * TT + t)) * DD + dcol] = f2bfu(O[f][r] * inv[r]);
    }
  }
}

// ---------------------------------------------------------------------------
// Kernel 3: out = O @ Wo + bo, bf16 in (Wo pre-transposed), f32 out.
// Same LDS-free structure as kernel 1.
// ---------------------------------------------------------------------------
__global__ __launch_bounds__(128) void out_proj_gemm(
    const unsigned short* __restrict__ a, const unsigned short* __restrict__ Wt,
    const float* __restrict__ bias, float* __restrict__ out)
{
  const int tid = threadIdx.x;
  const int wv = tid >> 5, ln = tid & 31;
  const int h2 = ln >> 4, nc = ln & 15;
  const int tileN = blockIdx.x * 64, tileM = blockIdx.y * 64;

  const unsigned short* arow = a + (size_t)(tileM + 16 * wv + nc) * DD;
  const unsigned short* brow0 = Wt + (size_t)(tileN + nc) * DD + 16 * h2;

  v8f acc[4] = {};

#pragma unroll 1
  for (int k0 = 0; k0 < DD; k0 += 32) {
    BFrag af;
    af.h[0] = *reinterpret_cast<const u16x8*>(&arow[k0 + 16 * h2]);
    af.h[1] = *reinterpret_cast<const u16x8*>(&arow[k0 + 16 * h2 + 8]);
#pragma unroll
    for (int f = 0; f < 4; ++f) {
      const unsigned short* b = brow0 + (size_t)(16 * f) * DD + k0;
      BFrag bfr;
      bfr.h[0] = *reinterpret_cast<const u16x8*>(b);
      bfr.h[1] = *reinterpret_cast<const u16x8*>(b + 8);
      acc[f] = __builtin_amdgcn_wmma_f32_16x16x32_bf16(
          false, af.v, false, bfr.v, (short)0, acc[f], false, false);
    }
  }

#pragma unroll
  for (int f = 0; f < 4; ++f) {
    const int n = tileN + 16 * f + nc;
    const float bv = bias[n];
#pragma unroll
    for (int r = 0; r < 8; ++r) {
      const int m = tileM + 16 * wv + r + 8 * h2;
      out[(size_t)m * DD + n] = acc[f][r] + bv;
    }
  }
}

// ---------------------------------------------------------------------------
extern "C" void kernel_launch(void* const* d_in, const int* in_sizes, int n_in,
                              void* d_out, int out_size, void* d_ws, size_t ws_size,
                              hipStream_t stream) {
  const float* x    = (const float*)d_in[0];
  // d_in[1] = mask (bool) -- causality implemented analytically, unused
  const float* cosT = (const float*)d_in[2];
  const float* sinT = (const float*)d_in[3];
  const float* Wq   = (const float*)d_in[4];
  const float* bq   = (const float*)d_in[5];
  const float* Wk   = (const float*)d_in[6];
  const float* bk   = (const float*)d_in[7];
  const float* Wv   = (const float*)d_in[8];
  const float* bv   = (const float*)d_in[9];
  const float* Wo   = (const float*)d_in[10];
  const float* bo   = (const float*)d_in[11];

  // workspace (bf16): q,k [B,H,T,HD]; v [B,H,HD,T]; o [B,T,D];
  // x_bf [B*T,D]; 4 transposed weights [D,D]  -> 88 MB total
  const size_t E  = (size_t)MM * DD;   // 8 Mi elements
  const size_t WE = (size_t)DD * DD;   // 1 Mi elements
  unsigned short* qws = (unsigned short*)d_ws;
  unsigned short* kws = qws + E;
  unsigned short* vws = kws + E;
  unsigned short* ows = vws + E;
  unsigned short* xbf = ows + E;
  unsigned short* wqT = xbf + E;
  unsigned short* wkT = wqT + WE;
  unsigned short* wvT = wkT + WE;
  unsigned short* woT = wvT + WE;

  dim3 blk128(128), blk256(256);
  dim3 gT(DD / 32, DD / 32);                    // weight transpose tiles
  convert_bf16<<<dim3((unsigned)(E / 1024)), blk256, 0, stream>>>(x, xbf);
  transpose_bf16<<<gT, blk256, 0, stream>>>(Wq, wqT);
  transpose_bf16<<<gT, blk256, 0, stream>>>(Wk, wkT);
  transpose_bf16<<<gT, blk256, 0, stream>>>(Wv, wvT);
  transpose_bf16<<<gT, blk256, 0, stream>>>(Wo, woT);

  dim3 gGemm(DD / 64, MM / 64);                 // 16 x 128 blocks
  qkv_rope_gemm<<<gGemm, blk128, 0, stream>>>(xbf, wqT, bq, cosT, sinT, qws, 1);
  qkv_rope_gemm<<<gGemm, blk128, 0, stream>>>(xbf, wkT, bk, cosT, sinT, kws, 1);
  qkv_rope_gemm<<<gGemm, blk128, 0, stream>>>(xbf, wvT, bv, cosT, sinT, vws, 0);
  flash_attn<<<dim3(TT / 64, BB * HH), blk128, 0, stream>>>(qws, kws, vws, ows);
  out_proj_gemm<<<gGemm, blk128, 0, stream>>>(ows, woT, bo, (float*)d_out);

  (void)in_sizes; (void)n_in; (void)out_size; (void)ws_size;
}
